// BinaryLlamaDecoderLayer_55731495633323
// MI455X (gfx1250) — matrix operand order repaired and checked by reading, not compiled, not run on hardware
//
#include <hip/hip_runtime.h>

// ---------------------------------------------------------------------------
// BinaryLlamaDecoderLayer for MI455X (gfx1250, wave32, WMMA)
// All GEMMs: v_wmma_f32_16x16x32_bf16 (binarized weights are +-aa -> exact bf16)
// GEMM tiles staged with GLOBAL_LOAD_ASYNC_TO_LDS_B128 (ASYNCcnt) when the
// toolchain exposes the builtin; double-buffered so DMA overlaps WMMA.
// Attention: fused flash-style, WMMA for QK^T and P*V, online softmax.
// ---------------------------------------------------------------------------

typedef __attribute__((ext_vector_type(16))) __bf16 v16bf;
typedef __attribute__((ext_vector_type(8)))  __bf16 v8bf;
typedef __attribute__((ext_vector_type(8)))  float  v8f;

// ---- async global->LDS path (gfx1250) ----
#if defined(__gfx1250__) && defined(__has_builtin)
#if __has_builtin(__builtin_amdgcn_global_load_async_to_lds_b128)
#define HAVE_ASYNC 1
#endif
#endif
#ifndef HAVE_ASYNC
#define HAVE_ASYNC 0
#endif

#if HAVE_ASYNC
// builtin signature (from hipcc diagnostic): (int4 AS1*, int4 AS3*, imm, imm)
typedef int i32x4 __attribute__((__vector_size__(4 * sizeof(int))));
typedef __attribute__((address_space(1))) i32x4 gvec_t;  // global int4
typedef __attribute__((address_space(3))) i32x4 lvec_t;  // LDS int4
__device__ __forceinline__ void async_copy_b128(const __bf16* g, __bf16* l) {
  // generic global ptr value == AS1 address; generic LDS ptr low 32 bits ==
  // LDS byte offset (aperture layout, ISA 10.2) -> integer casts are exact.
  gvec_t* gp = (gvec_t*)(unsigned long long)(size_t)g;
  lvec_t* lp = (lvec_t*)(unsigned)(size_t)l;
  __builtin_amdgcn_global_load_async_to_lds_b128(gp, lp, 0, 0);
}
#endif

__device__ __forceinline__ void wait_async_all() {
#if HAVE_ASYNC
#if __has_builtin(__builtin_amdgcn_s_wait_asynccnt)
  __builtin_amdgcn_s_wait_asynccnt(0);
#else
  asm volatile("s_wait_asynccnt 0" ::: "memory");
#endif
#endif
}

// ---- problem constants ----
constexpr int B_   = 2;
constexpr int S_   = 2048;
constexpr int H_   = 2048;
constexpr int NH_  = 32;
constexpr int NKV_ = 8;
constexpr int HD_  = 64;
constexpr int FF_  = 5632;
constexpr size_t MS = (size_t)B_ * S_;          // 4096 token rows

// ---- workspace layout (bytes) ----
constexpr size_t SZ_WQ = (size_t)H_ * H_ * 2;             // q/o weights bf16
constexpr size_t SZ_WK = (size_t)(NKV_*HD_) * H_ * 2;     // k/v weights bf16
constexpr size_t SZ_WG = (size_t)FF_ * H_ * 2;            // gate/up/down bf16
constexpr size_t OFF_WQ  = 0;
constexpr size_t OFF_WK  = OFF_WQ + SZ_WQ;
constexpr size_t OFF_WV  = OFF_WK + SZ_WK;
constexpr size_t OFF_WO  = OFF_WV + SZ_WK;
constexpr size_t OFF_WG  = OFF_WO + SZ_WQ;
constexpr size_t OFF_WU  = OFF_WG + SZ_WG;
constexpr size_t OFF_WD  = OFF_WU + SZ_WG;
constexpr size_t SZ_H1   = MS * H_ * 2;                   // normed activations bf16
constexpr size_t OFF_H1  = OFF_WD + SZ_WG;                // reused for h2 after attn
constexpr size_t SZ_QF   = MS * H_ * 4;                   // fp32 q proj (later: hidden fp32)
constexpr size_t OFF_QF  = OFF_H1 + SZ_H1;
constexpr size_t SZ_KF   = MS * (NKV_*HD_) * 4;
constexpr size_t OFF_KF  = OFF_QF + SZ_QF;
constexpr size_t OFF_VF  = OFF_KF + SZ_KF;
constexpr size_t SZ_QB   = MS * H_ * 2;                   // roped q bf16
constexpr size_t OFF_QB  = OFF_VF + SZ_KF;
constexpr size_t SZ_KB   = MS * (NKV_*HD_) * 2;
constexpr size_t OFF_KB  = OFF_QB + SZ_QB;
constexpr size_t OFF_VT  = OFF_KB + SZ_KB;                // V transposed [b,kvh,hd,S] bf16
constexpr size_t SZ_AO   = MS * H_ * 2;                   // attention output bf16
constexpr size_t OFF_AO  = OFF_VT + SZ_KB;
constexpr size_t OFF_ACT = OFF_AO + SZ_AO;                // silu(gate)*up bf16
// end = OFF_ACT + MS*FF_*2  ~= 246 MB

// ---------------------------------------------------------------------------
// helpers
// ---------------------------------------------------------------------------
__device__ __forceinline__ v8f vzero() {
  v8f z;
#pragma unroll
  for (int i = 0; i < 8; ++i) z[i] = 0.f;
  return z;
}

__device__ __forceinline__ v16bf make16(v8bf lo, v8bf hi) {
  v16bf r;
#pragma unroll
  for (int i = 0; i < 8; ++i) { r[i] = lo[i]; r[i + 8] = hi[i]; }
  return r;
}

// A-matrix fragment (16x32, 16-bit): lane row = lane&15,
// lanes 0-15 hold K {0..7,16..23}; lanes 16-31 hold K {8..15,24..31}.
__device__ __forceinline__ v16bf afrag(const __bf16* rowp, int lane) {
  const int kb = (lane & 16) ? 8 : 0;
  return make16(*(const v8bf*)(rowp + kb), *(const v8bf*)(rowp + kb + 16));
}

// B-matrix fragment (32x16, 16-bit): lane col = lane&15,
// lanes 0-15 hold K {0..7,16..23}; lanes 16-31 hold K {8..15,24..31}.
__device__ __forceinline__ v16bf bfrag(const __bf16* colp, int lane) {
  const int kb = (lane & 16) ? 8 : 0;
  return make16(*(const v8bf*)(colp + kb), *(const v8bf*)(colp + kb + 16));
}

__device__ __forceinline__ v8f wmma_bf16(v16bf a, v16bf b, v8f c) {
  return __builtin_amdgcn_wmma_f32_16x16x32_bf16(false, a, false, b, (short)0, c,
                                                 false, false);
}

// reductions within a 16-lane half of the wave (xor masks stay inside halves)
__device__ __forceinline__ float hmax16(float v) {
#pragma unroll
  for (int o = 8; o >= 1; o >>= 1) v = fmaxf(v, __shfl_xor(v, o));
  return v;
}
__device__ __forceinline__ float hsum16(float v) {
#pragma unroll
  for (int o = 8; o >= 1; o >>= 1) v += __shfl_xor(v, o);
  return v;
}

// ---------------------------------------------------------------------------
// elementwise kernels
// ---------------------------------------------------------------------------
__global__ __launch_bounds__(256) void binarize_kernel(
    const float* __restrict__ W, __bf16* __restrict__ O,
    const float* __restrict__ kk, const float* __restrict__ aa, int n) {
  int i = blockIdx.x * 256 + threadIdx.x;
  if (i >= n) return;
  float w = kk[0] * W[i];
  w = fminf(1.f, fmaxf(-1.f, w));
  O[i] = (__bf16)(aa[0] * w);
}

__global__ __launch_bounds__(256) void rmsnorm_kernel(
    const float* __restrict__ X, const float* __restrict__ W,
    __bf16* __restrict__ O) {
  __shared__ float red[8];
  const int row = blockIdx.x, tid = threadIdx.x;
  const float* xr = X + (size_t)row * H_;
  float ss = 0.f;
  for (int i = tid; i < H_; i += 256) { float v = xr[i]; ss += v * v; }
#pragma unroll
  for (int o = 16; o >= 1; o >>= 1) ss += __shfl_xor(ss, o);
  if ((tid & 31) == 0) red[tid >> 5] = ss;
  __syncthreads();
  float tot = 0.f;
#pragma unroll
  for (int i = 0; i < 8; ++i) tot += red[i];
  const float inv = rsqrtf(tot * (1.f / H_) + 1e-5f);
  __bf16* orow = O + (size_t)row * H_;
  for (int i = tid; i < H_; i += 256) orow[i] = (__bf16)(xr[i] * inv * W[i]);
}

__global__ __launch_bounds__(256) void rope_kernel(
    const float* __restrict__ X, __bf16* __restrict__ O,
    const int* __restrict__ pos, int nh, int n) {
  int i = blockIdx.x * 256 + threadIdx.x;
  if (i >= n) return;
  const int d  = i & 31;
  const int h  = (i >> 5) % nh;
  const int bs = i / (32 * nh);
  const float p   = (float)pos[bs];
  const float inv = __powf(10000.f, -(float)(2 * d) * (1.f / 64.f));
  float sn, cs;
  __sincosf(p * inv, &sn, &cs);
  const size_t base = (size_t)bs * (nh * 64) + h * 64 + d;
  const float x1 = X[base], x2 = X[base + 32];
  O[base]      = (__bf16)(x1 * cs - x2 * sn);
  O[base + 32] = (__bf16)(x2 * cs + x1 * sn);
}

// V: [b,s,kvh,hd] fp32 -> [b,kvh,hd,S] bf16 (K-contiguous for WMMA B-frags)
__global__ __launch_bounds__(256) void packv_kernel(
    const float* __restrict__ X, __bf16* __restrict__ O, int n) {
  int i = blockIdx.x * 256 + threadIdx.x;
  if (i >= n) return;
  const int d  = i & 63;
  const int h  = (i >> 6) & (NKV_ - 1);
  const int bs = i >> 9;
  const int b = bs / S_, s = bs % S_;
  O[(((size_t)(b * NKV_ + h)) * HD_ + d) * S_ + s] = (__bf16)X[i];
}

// ---------------------------------------------------------------------------
// tiled GEMM: C[M,N] = A[M,K] * W[N,K]^T  (A,W bf16; acc fp32)
// block = 256 threads = 8 waves (4x2); block tile 128x64, BK=32.
// Double-buffered LDS; tiles staged via async global->LDS DMA when available,
// prefetch for step i+1 overlaps the WMMA work of step i.
// EPI: 0 -> store fp32 | 1 -> store fp32 + residual | 2 -> dual-B silu(g)*u bf16
// ---------------------------------------------------------------------------
template <int EPI>
__global__ __launch_bounds__(256) void gemm_bt_kernel(
    const __bf16* __restrict__ A, const __bf16* __restrict__ B0,
    const __bf16* __restrict__ B1, const float* __restrict__ Res,
    float* __restrict__ Cf, __bf16* __restrict__ Cb, int M, int N, int K) {
  constexpr int LDT = 40;  // 32 + 8 pad halves (80B rows, 16B aligned)
  __shared__ __bf16 sA[2][128 * LDT];
  __shared__ __bf16 sB[2][64 * LDT];
  __shared__ __bf16 sB1[(EPI == 2) ? 2 * 64 * LDT : 2];

  const int tid = threadIdx.x, lane = tid & 31, wave = tid >> 5;
  const int wm = wave >> 1, wn = wave & 1;
  const int mblock = blockIdx.y * 128, nblock = blockIdx.x * 64;
  const int ln = lane & 15, hi2 = lane >> 4;

  v8f acc[2][2], acc2[2][2];
#pragma unroll
  for (int i = 0; i < 2; ++i)
#pragma unroll
    for (int j = 0; j < 2; ++j) { acc[i][j] = vzero(); acc2[i][j] = vzero(); }

  const int arow = tid >> 1, akc = (tid & 1) * 16;  // 128 rows x 32 halves
  const int brow = tid >> 2, bkc = (tid & 3) * 8;   // 64 rows x 32 halves

  auto stage = [&](int kt, int buf) {
    const __bf16* ga = A + (size_t)(mblock + arow) * K + kt + akc;
    __bf16* la = &sA[buf][arow * LDT + akc];
    const __bf16* gb = B0 + (size_t)(nblock + brow) * K + kt + bkc;
    __bf16* lb = &sB[buf][brow * LDT + bkc];
#if HAVE_ASYNC
    async_copy_b128(ga, la);
    async_copy_b128(ga + 8, la + 8);
    async_copy_b128(gb, lb);
#else
    *(v8bf*)la       = *(const v8bf*)ga;
    *(v8bf*)(la + 8) = *(const v8bf*)(ga + 8);
    *(v8bf*)lb       = *(const v8bf*)gb;
#endif
    if (EPI == 2) {
      const __bf16* gb1 = B1 + (size_t)(nblock + brow) * K + kt + bkc;
      __bf16* lb1 = &sB1[buf * 64 * LDT + brow * LDT + bkc];
#if HAVE_ASYNC
      async_copy_b128(gb1, lb1);
#else
      *(v8bf*)lb1 = *(const v8bf*)gb1;
#endif
    }
  };

  stage(0, 0);
  const int nk = K / 32;
  for (int ik = 0; ik < nk; ++ik) {
    const int cur = ik & 1;
    wait_async_all();      // this wave's DMA into `cur` has landed in LDS
    __syncthreads();       // every wave's `cur` slice ready; `nxt` readers done
    if (ik + 1 < nk) stage((ik + 1) * 32, cur ^ 1);  // overlaps WMMA below

    const v16bf a0 = afrag(&sA[cur][(wm * 32 + ln) * LDT], lane);
    const v16bf a1 = afrag(&sA[cur][(wm * 32 + 16 + ln) * LDT], lane);
    const v16bf b0 = bfrag(&sB[cur][(wn * 32 + ln) * LDT], lane);
    const v16bf b1 = bfrag(&sB[cur][(wn * 32 + 16 + ln) * LDT], lane);
    acc[0][0] = wmma_bf16(a0, b0, acc[0][0]);
    acc[0][1] = wmma_bf16(a0, b1, acc[0][1]);
    acc[1][0] = wmma_bf16(a1, b0, acc[1][0]);
    acc[1][1] = wmma_bf16(a1, b1, acc[1][1]);
    if (EPI == 2) {
      const v16bf c0 = bfrag(&sB1[cur * 64 * LDT + (wn * 32 + ln) * LDT], lane);
      const v16bf c1 = bfrag(&sB1[cur * 64 * LDT + (wn * 32 + 16 + ln) * LDT], lane);
      acc2[0][0] = wmma_bf16(a0, c0, acc2[0][0]);
      acc2[0][1] = wmma_bf16(a0, c1, acc2[0][1]);
      acc2[1][0] = wmma_bf16(a1, c0, acc2[1][0]);
      acc2[1][1] = wmma_bf16(a1, c1, acc2[1][1]);
    }
  }

  // C layout: VGPR v, lanes<16 -> row v; lanes>=16 -> row v+8; col = lane&15
#pragma unroll
  for (int i = 0; i < 2; ++i)
#pragma unroll
    for (int j = 0; j < 2; ++j) {
#pragma unroll
      for (int v = 0; v < 8; ++v) {
        const int row = mblock + wm * 32 + i * 16 + v + 8 * hi2;
        const int col = nblock + wn * 32 + j * 16 + ln;
        const size_t idx = (size_t)row * N + col;
        const float val = acc[i][j][v];
        if (EPI == 0) {
          Cf[idx] = val;
        } else if (EPI == 1) {
          Cf[idx] = val + Res[idx];
        } else {
          const float u  = acc2[i][j][v];
          const float sg = val / (1.f + __expf(-val));
          Cb[idx] = (__bf16)(sg * u);
        }
      }
    }
}

// ---------------------------------------------------------------------------
// fused causal GQA flash attention (bf16 WMMA, fp32 online softmax)
// grid.x = B*NH*(S/128); block = 256 = 8 waves; each wave owns a 16-row Q tile
// ---------------------------------------------------------------------------
__global__ __launch_bounds__(256) void flash_attn_kernel(
    const __bf16* __restrict__ Q, const __bf16* __restrict__ Kc,
    const __bf16* __restrict__ Vt, __bf16* __restrict__ O) {
  __shared__ __bf16 pls[8][16 * 40];  // per-wave P bounce buffer (16x32, pad 40)
  const int tid = threadIdx.x, lane = tid & 31, wave = tid >> 5;
  const int ln = lane & 15, hi2 = lane >> 4;
  constexpr int TILES = S_ / 128;
  const int t128 = blockIdx.x % TILES;
  const int bh   = blockIdx.x / TILES;
  const int b = bh >> 5, h = bh & 31, kvh = h >> 2;
  const int s0 = t128 * 128 + wave * 16;

  const __bf16* qrow = Q + ((size_t)(b * S_ + s0 + ln) * (NH_ * HD_) + h * HD_);
  const v16bf qf0 = afrag(qrow, lane);       // hd 0..31
  const v16bf qf1 = afrag(qrow + 32, lane);  // hd 32..63

  v8f acc[4];
#pragma unroll
  for (int n = 0; n < 4; ++n) acc[n] = vzero();
  float mr[8], lr[8];
#pragma unroll
  for (int j = 0; j < 8; ++j) { mr[j] = -1e30f; lr[j] = 0.f; }

  const int kvend = s0 + 16;  // causal: no KV beyond this wave's diagonal
  for (int kt = 0; kt < kvend; kt += 32) {
    // --- scores: 16(q) x 32(kv), K-dim = HD = 64 -> 2 wmma per 16-col tile ---
    v8f sc[2];
#pragma unroll
    for (int t = 0; t < 2; ++t) {
      const __bf16* kcol =
          Kc + ((size_t)(b * S_ + kt + t * 16 + ln) * (NKV_ * HD_) + kvh * HD_);
      const v16bf kf0 = bfrag(kcol, lane);
      const v16bf kf1 = bfrag(kcol + 32, lane);
      v8f c = vzero();
      c = wmma_bf16(qf0, kf0, c);
      c = wmma_bf16(qf1, kf1, c);
      sc[t] = c;
    }
    // --- scale, causal mask, online softmax stats ---
    float alpha[8];
#pragma unroll
    for (int j = 0; j < 8; ++j) {
      const int row = s0 + j + 8 * hi2;
      const int c0 = kt + ln, c1 = kt + 16 + ln;
      const float x0 = (c0 <= row) ? sc[0][j] * 0.125f : -1e30f;
      const float x1 = (c1 <= row) ? sc[1][j] * 0.125f : -1e30f;
      const float rm = hmax16(fmaxf(x0, x1));
      const float mnew = fmaxf(mr[j], rm);
      const float a  = __expf(mr[j] - mnew);
      const float p0 = __expf(x0 - mnew);
      const float p1 = __expf(x1 - mnew);
      lr[j] = lr[j] * a + hsum16(p0 + p1);
      mr[j] = mnew;
      alpha[j] = a;
      pls[wave][(j + 8 * hi2) * 40 + ln]      = (__bf16)p0;
      pls[wave][(j + 8 * hi2) * 40 + 16 + ln] = (__bf16)p1;
    }
#pragma unroll
    for (int n = 0; n < 4; ++n)
#pragma unroll
      for (int j = 0; j < 8; ++j) acc[n][j] *= alpha[j];

    // LDS RAW across lanes of this wave: DS ops are in-order per wave; the
    // wait + memory clobber keeps the compiler from hoisting the reads.
    asm volatile("s_wait_dscnt 0" ::: "memory");
    const v16bf pf = afrag(&pls[wave][ln * 40], lane);  // P as A-frag (16x32)
#pragma unroll
    for (int n = 0; n < 4; ++n) {
      const __bf16* vcol =
          Vt + (((size_t)(b * NKV_ + kvh) * HD_ + n * 16 + ln) * S_ + kt);
      const v16bf vf = bfrag(vcol, lane);  // kv-contiguous
      acc[n] = wmma_bf16(pf, vf, acc[n]);
    }
  }

#pragma unroll
  for (int n = 0; n < 4; ++n)
#pragma unroll
    for (int j = 0; j < 8; ++j) {
      const int row = s0 + j + 8 * hi2;
      const int col = n * 16 + ln;
      O[(size_t)(b * S_ + row) * (NH_ * HD_) + h * HD_ + col] =
          (__bf16)(acc[n][j] / lr[j]);
    }
}

// ---------------------------------------------------------------------------
// launch
// ---------------------------------------------------------------------------
extern "C" void kernel_launch(void* const* d_in, const int* in_sizes, int n_in,
                              void* d_out, int out_size, void* d_ws, size_t ws_size,
                              hipStream_t stream) {
  (void)in_sizes; (void)n_in; (void)out_size; (void)ws_size;
  const float* hidden = (const float*)d_in[0];
  const int*   pos    = (const int*)d_in[2];
  const float* qw = (const float*)d_in[3];
  const float* kw = (const float*)d_in[4];
  const float* vw = (const float*)d_in[5];
  const float* ow = (const float*)d_in[6];
  const float* gw = (const float*)d_in[7];
  const float* uw = (const float*)d_in[8];
  const float* dw = (const float*)d_in[9];
  const float* ln1 = (const float*)d_in[10];
  const float* ln2 = (const float*)d_in[11];
  const float* kk  = (const float*)d_in[12];
  const float* aa  = (const float*)d_in[13];

  char* ws = (char*)d_ws;
  __bf16* WQ  = (__bf16*)(ws + OFF_WQ);
  __bf16* WK  = (__bf16*)(ws + OFF_WK);
  __bf16* WV  = (__bf16*)(ws + OFF_WV);
  __bf16* WO  = (__bf16*)(ws + OFF_WO);
  __bf16* WG  = (__bf16*)(ws + OFF_WG);
  __bf16* WU  = (__bf16*)(ws + OFF_WU);
  __bf16* WD  = (__bf16*)(ws + OFF_WD);
  __bf16* H1  = (__bf16*)(ws + OFF_H1);   // rmsnorm output (reused for h2)
  float*  QF  = (float*)(ws + OFF_QF);    // fp32 q proj; later hidden fp32
  float*  KF  = (float*)(ws + OFF_KF);
  float*  VF  = (float*)(ws + OFF_VF);
  __bf16* QB  = (__bf16*)(ws + OFF_QB);
  __bf16* KB  = (__bf16*)(ws + OFF_KB);
  __bf16* VT  = (__bf16*)(ws + OFF_VT);
  __bf16* AO  = (__bf16*)(ws + OFF_AO);
  __bf16* ACT = (__bf16*)(ws + OFF_ACT);
  float*  HIDDENF = QF;  // reuse after RoPE consumed QF
  float*  OUT = (float*)d_out;

  const dim3 blk(256);

  // 1) binarize weights -> bf16
  binarize_kernel<<<(H_ * H_) / 256, blk, 0, stream>>>(qw, WQ, kk, aa, H_ * H_);
  binarize_kernel<<<(NKV_ * HD_ * H_) / 256, blk, 0, stream>>>(kw, WK, kk, aa, NKV_ * HD_ * H_);
  binarize_kernel<<<(NKV_ * HD_ * H_) / 256, blk, 0, stream>>>(vw, WV, kk, aa, NKV_ * HD_ * H_);
  binarize_kernel<<<(H_ * H_) / 256, blk, 0, stream>>>(ow, WO, kk, aa, H_ * H_);
  binarize_kernel<<<(FF_ * H_) / 256, blk, 0, stream>>>(gw, WG, kk, aa, FF_ * H_);
  binarize_kernel<<<(FF_ * H_) / 256, blk, 0, stream>>>(uw, WU, kk, aa, FF_ * H_);
  binarize_kernel<<<(FF_ * H_) / 256, blk, 0, stream>>>(dw, WD, kk, aa, FF_ * H_);

  // 2) rmsnorm 1
  rmsnorm_kernel<<<(int)MS, blk, 0, stream>>>(hidden, ln1, H1);

  // 3) q/k/v projections (WMMA)
  gemm_bt_kernel<0><<<dim3(H_ / 64, MS / 128), blk, 0, stream>>>(
      H1, WQ, nullptr, nullptr, QF, nullptr, (int)MS, H_, H_);
  gemm_bt_kernel<0><<<dim3((NKV_ * HD_) / 64, MS / 128), blk, 0, stream>>>(
      H1, WK, nullptr, nullptr, KF, nullptr, (int)MS, NKV_ * HD_, H_);
  gemm_bt_kernel<0><<<dim3((NKV_ * HD_) / 64, MS / 128), blk, 0, stream>>>(
      H1, WV, nullptr, nullptr, VF, nullptr, (int)MS, NKV_ * HD_, H_);

  // 4) RoPE + pack to bf16; V -> transposed layout
  {
    const int nq = (int)(MS * H_ / 2);             // pairs
    rope_kernel<<<nq / 256, blk, 0, stream>>>(QF, QB, pos, NH_, nq);
    const int nk = (int)(MS * NKV_ * HD_ / 2);
    rope_kernel<<<nk / 256, blk, 0, stream>>>(KF, KB, pos, NKV_, nk);
    const int nv = (int)(MS * NKV_ * HD_);
    packv_kernel<<<nv / 256, blk, 0, stream>>>(VF, VT, nv);
  }

  // 5) fused causal flash attention
  flash_attn_kernel<<<B_ * NH_ * (S_ / 128), blk, 0, stream>>>(QB, KB, VT, AO);

  // 6) o projection + residual -> hidden fp32
  gemm_bt_kernel<1><<<dim3(H_ / 64, MS / 128), blk, 0, stream>>>(
      AO, WO, nullptr, hidden, HIDDENF, nullptr, (int)MS, H_, H_);

  // 7) rmsnorm 2
  rmsnorm_kernel<<<(int)MS, blk, 0, stream>>>(HIDDENF, ln2, H1);

  // 8) fused gate/up GEMM with silu(g)*u epilogue -> bf16
  gemm_bt_kernel<2><<<dim3(FF_ / 64, MS / 128), blk, 0, stream>>>(
      H1, WG, WU, nullptr, nullptr, ACT, (int)MS, FF_, H_);

  // 9) down projection + residual -> output fp32
  gemm_bt_kernel<1><<<dim3(H_ / 64, MS / 128), blk, 0, stream>>>(
      ACT, WD, nullptr, HIDDENF, OUT, nullptr, (int)MS, H_, FF_);
}
